// HyperbolicAttention_41317585387571
// MI455X (gfx1250) — matrix operand order repaired and checked by reading, not compiled, not run on hardware
//
#include <hip/hip_runtime.h>
#include <math.h>

typedef __attribute__((ext_vector_type(16))) _Float16 v16h;
typedef __attribute__((ext_vector_type(8)))  _Float16 v8h;
typedef __attribute__((ext_vector_type(8)))  float    v8f;

#define BH_TOT 64
#define SEQ_N  1024
#define DIM    64
#define BM     128   // query rows per workgroup (8 waves x 16)
#define BN     32    // key block
#define NWAVES 8
#define EPSF   1e-6f

__global__ __launch_bounds__(256)
void hyp_attn_kernel(const float* __restrict__ Q, const float* __restrict__ K,
                     const float* __restrict__ V, const float* __restrict__ betaP,
                     const float* __restrict__ cP, float* __restrict__ Out)
{
    __shared__ alignas(32) _Float16 Klds[BN * DIM];          // [key][dim]  f16, 4KB
    __shared__ alignas(32) _Float16 Vtlds[DIM * BN];         // [dim][key]  f16, 4KB (transposed)
    __shared__ alignas(32) float    k0lds[BN];
    __shared__ alignas(32) _Float16 v0lds[BN];
    __shared__ alignas(32) float    q0lds[NWAVES * 16];
    __shared__ alignas(32) _Float16 albuf[NWAVES * 16 * BN]; // per-wave alpha scratch, 8KB

    const int tid  = threadIdx.x;
    const int lane = tid & 31;
    const int wave = tid >> 5;
    const int h16  = lane >> 4;     // 0/1: which half of the wave
    const int l16  = lane & 15;
    const int h8   = h16 * 8;
    const int bh   = blockIdx.y;
    const int qbase = blockIdx.x * BM + wave * 16;

    const float beta = betaP[0];
    const float cl2  = cP[0] * 1.44269504088896340736f;   // c * log2(e)

    // ---------- preload Q A-fragments (two K=32 chunks) + q0 ----------
    const float* qrow = Q + ((size_t)bh * SEQ_N + (qbase + l16)) * DIM;
    v16h aQ[2];
    float sq = 0.f;
    #pragma unroll
    for (int c = 0; c < 2; ++c) {
        float4 g0 = *(const float4*)(qrow + c * 32 + h16 * 16);
        float4 g1 = *(const float4*)(qrow + c * 32 + h16 * 16 + 4);
        float4 g2 = *(const float4*)(qrow + c * 32 + h16 * 16 + 8);
        float4 g3 = *(const float4*)(qrow + c * 32 + h16 * 16 + 12);
        float f[16] = {g0.x,g0.y,g0.z,g0.w, g1.x,g1.y,g1.z,g1.w,
                       g2.x,g2.y,g2.z,g2.w, g3.x,g3.y,g3.z,g3.w};
        #pragma unroll
        for (int i = 0; i < 16; ++i) { aQ[c][i] = (_Float16)f[i]; sq += f[i]*f[i]; }
    }
    sq += __shfl_xor(sq, 16, 32);                  // lane L + L^16 cover the full row
    if (lane < 16) q0lds[wave * 16 + l16] = __builtin_amdgcn_sqrtf(1.f + sq);
    asm volatile("" ::: "memory");                 // same-wave LDS is in-order (ISA 7.3)
    float q0r[8];
    #pragma unroll
    for (int r = 0; r < 8; ++r) q0r[r] = q0lds[wave * 16 + h8 + r];  // row M = r + 8*h16

    v8f o[4], dn;
    #pragma unroll
    for (int t = 0; t < 4; ++t)
        #pragma unroll
        for (int r = 0; r < 8; ++r) o[t][r] = 0.f;
    #pragma unroll
    for (int r = 0; r < 8; ++r) dn[r] = 0.f;

    for (int kb = 0; kb < SEQ_N; kb += BN) {
        __syncthreads();   // previous block's LDS reads done
        // ---------- cooperative K/V block load, f32 -> f16, + k0/v0 ----------
        {
            const int row = tid >> 3;   // 0..31 key row
            const int seg = tid & 7;    // 8 floats each
            const float* kp = K + ((size_t)bh * SEQ_N + kb + row) * DIM + seg * 8;
            float4 a0 = *(const float4*)kp;
            float4 a1 = *(const float4*)(kp + 4);
            float f[8] = {a0.x,a0.y,a0.z,a0.w,a1.x,a1.y,a1.z,a1.w};
            float p = 0.f; v8h hk;
            #pragma unroll
            for (int i = 0; i < 8; ++i) { p += f[i]*f[i]; hk[i] = (_Float16)f[i]; }
            *(v8h*)(&Klds[row * DIM + seg * 8]) = hk;
            p += __shfl_xor(p, 1, 32); p += __shfl_xor(p, 2, 32); p += __shfl_xor(p, 4, 32);
            if (seg == 0) k0lds[row] = __builtin_amdgcn_sqrtf(1.f + p);

            const float* vp = V + ((size_t)bh * SEQ_N + kb + row) * DIM + seg * 8;
            float4 b0 = *(const float4*)vp;
            float4 b1 = *(const float4*)(vp + 4);
            float g[8] = {b0.x,b0.y,b0.z,b0.w,b1.x,b1.y,b1.z,b1.w};
            float pv = 0.f;
            #pragma unroll
            for (int i = 0; i < 8; ++i) {
                pv += g[i]*g[i];
                Vtlds[(seg * 8 + i) * BN + row] = (_Float16)g[i];   // transpose into LDS
            }
            pv += __shfl_xor(pv, 1, 32); pv += __shfl_xor(pv, 2, 32); pv += __shfl_xor(pv, 4, 32);
            if (seg == 0) v0lds[row] = (_Float16)__builtin_amdgcn_sqrtf(1.f + pv);
        }
        __syncthreads();

        // ---------- S = Q_spatial * K^T (two 16x16 tiles over 32 keys) ----------
        v8f s[2];
        #pragma unroll
        for (int t = 0; t < 2; ++t) {
            const _Float16* br = &Klds[(t * 16 + l16) * DIM + h16 * 16];
            v16h bc0 = *(const v16h*)(br);        // kdims 0..31 (per-lane half)
            v16h bc1 = *(const v16h*)(br + 32);   // kdims 32..63
            v8f acc;
            #pragma unroll
            for (int r = 0; r < 8; ++r) acc[r] = 0.f;
            acc = __builtin_amdgcn_wmma_f32_16x16x32_f16(false, aQ[0], false, bc0,
                                                         (short)0, acc, false, false);
            acc = __builtin_amdgcn_wmma_f32_16x16x32_f16(false, aQ[1], false, bc1,
                                                         (short)0, acc, false, false);
            s[t] = acc;
        }

        // ---------- alpha = exp2(-beta*log2(x+sqrt(x^2-1)) - c*log2e) ----------
        // x >= 1+eps everywhere -> raw HW sqrt/log/exp are safe (no denorm/range fixups)
        const float k0a = k0lds[l16];
        const float k0b = k0lds[16 + l16];
        _Float16* aw = &albuf[wave * 16 * BN];
        #pragma unroll
        for (int r = 0; r < 8; ++r) {
            #pragma unroll
            for (int t = 0; t < 2; ++t) {
                float x = fmaf(q0r[r], (t ? k0b : k0a), -s[t][r]);  // -<q,k>_L
                x = fmaxf(x, 1.f + EPSF);
                float tt = x + __builtin_amdgcn_sqrtf(fmaf(x, x, -1.f));
                float al = __builtin_amdgcn_exp2f(
                               fmaf(-beta, __builtin_amdgcn_logf(tt), -cl2));
                aw[(h8 + r) * BN + t * 16 + l16] = (_Float16)al;    // C/D -> row-major
            }
        }
        asm volatile("" ::: "memory");  // same-wave LDS in-order; just block reordering

        // ---------- repack alpha into A-fragment (16x32 f16) ----------
        v16h aA;
        {
            const _Float16* ar = &albuf[wave * 16 * BN + l16 * BN];
            v8h lo = *(const v8h*)(ar + h16 * 16);      // K = 16*h16 .. 16*h16+7
            v8h hi = *(const v8h*)(ar + h16 * 16 + 8);  // K = 16*h16+8 .. 16*h16+15
            #pragma unroll
            for (int i = 0; i < 8; ++i) { aA[i] = lo[i]; aA[8 + i] = hi[i]; }
        }

        // ---------- O += alpha * V (4 tiles), den += alpha * v0 ----------
        #pragma unroll
        for (int t = 0; t < 4; ++t) {
            v16h bv = *(const v16h*)(&Vtlds[(t * 16 + l16) * BN + h16 * 16]);
            o[t] = __builtin_amdgcn_wmma_f32_16x16x32_f16(false, aA, false, bv,
                                                          (short)0, o[t], false, false);
        }
        {
            v16h b0v = *(const v16h*)(&v0lds[h16 * 16]);  // broadcast v0 over all 16 cols
            dn = __builtin_amdgcn_wmma_f32_16x16x32_f16(false, aA, false, b0v,
                                                        (short)0, dn, false, false);
        }
    }

    // ---------- out = O / den (den tile layout matches O tiles lane-for-lane) ----------
    float rdn[8];
    #pragma unroll
    for (int r = 0; r < 8; ++r) rdn[r] = __builtin_amdgcn_rcpf(dn[r]);
    #pragma unroll
    for (int t = 0; t < 4; ++t)
        #pragma unroll
        for (int r = 0; r < 8; ++r) {
            const int row = qbase + h8 + r;
            Out[((size_t)bh * SEQ_N + row) * DIM + t * 16 + l16] = o[t][r] * rdn[r];
        }
}

extern "C" void kernel_launch(void* const* d_in, const int* in_sizes, int n_in,
                              void* d_out, int out_size, void* d_ws, size_t ws_size,
                              hipStream_t stream) {
    const float* q    = (const float*)d_in[0];
    const float* k    = (const float*)d_in[1];
    const float* v    = (const float*)d_in[2];
    const float* beta = (const float*)d_in[3];
    const float* c    = (const float*)d_in[4];
    float* out = (float*)d_out;
    dim3 grid(SEQ_N / BM, BH_TOT);
    hipLaunchKernelGGL(hyp_attn_kernel, grid, dim3(256), 0, stream, q, k, v, beta, c, out);
}
